// StudentMoEEncoder_47004122087600
// MI455X (gfx1250) — matrix operand order repaired and checked by reading, not compiled, not run on hardware
//
#include <hip/hip_runtime.h>
#include <hip/hip_bf16.h>
#include <math.h>

// Problem dims (match reference)
#define B_SZ  32768
#define D_SZ  512
#define H1_SZ 512
#define H2_SZ 256
#define E_SZ  8
#define L_SZ  32

#define TILE_M 64
#define LDA_OBS (D_SZ + 8)    // 520 (rows stay 16B-aligned, staggers banks)
#define LDA_X1  (H1_SZ + 8)   // 520
#define LDA_X2  (H2_SZ + 8)   // 264

typedef __attribute__((ext_vector_type(16))) __bf16 v16bf;
typedef __attribute__((ext_vector_type(8)))  float  v8f;
typedef __attribute__((ext_vector_type(4)))  int    v4i;

#if defined(__HIP_DEVICE_COMPILE__) && __has_builtin(__builtin_amdgcn_global_load_async_to_lds_b128)
#define HAVE_ASYNC_LDS 1
typedef __attribute__((address_space(1))) v4i* gptr_v4i;   // global int4*
typedef __attribute__((address_space(3))) v4i* lptr_v4i;   // LDS int4*
#else
#define HAVE_ASYNC_LDS 0
#endif

// ---------------------------------------------------------------------------
// A-fragment (bf16 16x32, MxK) from LDS.
// Lane {half,lm}: row m = lm; regs 0..3 hold K = half*8 + [0..8),
// regs 4..7 hold K = 16 + half*8 + [0..8) -> two ds_load_b128.
// ---------------------------------------------------------------------------
__device__ __forceinline__ v16bf load_a_frag(const __bf16* sA, int lda,
                                             int mRow, int kBase, int half) {
  union { v16bf v; uint4 q[2]; } u;
  const __bf16* p = sA + mRow * lda + kBase + half * 8;
  u.q[0] = *reinterpret_cast<const uint4*>(p);
  u.q[1] = *reinterpret_cast<const uint4*>(p + 16);
  return u.v;
}

// ---------------------------------------------------------------------------
// B-fragment (bf16 32x16, KxN) from global; weights stored COLUMN-major
// (Wt[n][k], row stride K). Lane: col n = lm, K = half*8 + [0..8) and
// 16 + half*8 + [0..8) -> two global_load_b128 (L2-resident weights).
// ---------------------------------------------------------------------------
__device__ __forceinline__ v16bf load_b_frag(const __bf16* __restrict__ Bt,
                                             int K, int nCol, int kBase, int half) {
  union { v16bf v; uint4 q[2]; } u;
  const __bf16* p = Bt + (size_t)nCol * K + kBase + half * 8;
  u.q[0] = *reinterpret_cast<const uint4*>(p);
  u.q[1] = *reinterpret_cast<const uint4*>(p + 16);
  return u.v;
}

// Branchless ELU: one v_exp_f32 (co-executes with WMMA) + cndmask.
__device__ __forceinline__ float elu_fast(float x) {
  float en = __expf(x) - 1.f;
  return (x > 0.f) ? x : en;
}

// ---------------------------------------------------------------------------
// GEMM phase: C = ELU(A @ B + bias) -> LDS bf16.
// A-fragments hoisted to registers once (K/32 frags); inner loop is pure
// global_load_b128 + v_wmma.
// ---------------------------------------------------------------------------
template <int K>
__device__ __forceinline__ void gemm_phase(const __bf16* sA, int lda,
                                           const __bf16* __restrict__ Bt,
                                           const float* __restrict__ bias,
                                           __bf16* sOut, int ldo,
                                           int mBase, int nStart, int nEnd,
                                           int half, int lm) {
  constexpr int KT = K / 32;
  v16bf afrag[KT];
#pragma unroll
  for (int i = 0; i < KT; ++i)
    afrag[i] = load_a_frag(sA, lda, mBase + lm, i * 32, half);

  for (int nBase = nStart; nBase < nEnd; nBase += 16) {
    const int n = nBase + lm;
    const float bv = bias[n];        // issued early, overlaps with B loads
    v8f acc = {0.f, 0.f, 0.f, 0.f, 0.f, 0.f, 0.f, 0.f};
#pragma unroll
    for (int i = 0; i < KT; ++i) {
      v16bf b = load_b_frag(Bt, K, n, i * 32, half);
      acc = __builtin_amdgcn_wmma_f32_16x16x32_bf16(
          false, afrag[i], false, b, (short)0, acc, false, false);
    }
    const int m0 = mBase + half * 8;   // C/D layout: elem r -> (m0+r, n)
#pragma unroll
    for (int r = 0; r < 8; ++r)
      sOut[(m0 + r) * ldo + n] = (__bf16)elu_fast(acc[r] + bv);
  }
}

// ---------------------------------------------------------------------------
// Expert head: lat = x2 @ W3 + b3 (no activation), then
// sLat[m, n] += gate[m, e] * lat.  Each wave owns a unique 16x16 tile.
// ---------------------------------------------------------------------------
__device__ __forceinline__ void gemm3_phase(const __bf16* sX2,
                                            const __bf16* __restrict__ Bt,
                                            const float* __restrict__ bias,
                                            const float* sGate, float* sLat,
                                            int e, int mBase, int nBase,
                                            int half, int lm) {
  constexpr int KT = H2_SZ / 32;
  v16bf afrag[KT];
#pragma unroll
  for (int i = 0; i < KT; ++i)
    afrag[i] = load_a_frag(sX2, LDA_X2, mBase + lm, i * 32, half);

  const int n = nBase + lm;
  const float bv = bias[n];
  v8f acc = {0.f, 0.f, 0.f, 0.f, 0.f, 0.f, 0.f, 0.f};
#pragma unroll
  for (int i = 0; i < KT; ++i) {
    v16bf b = load_b_frag(Bt, H2_SZ, n, i * 32, half);
    acc = __builtin_amdgcn_wmma_f32_16x16x32_bf16(
        false, afrag[i], false, b, (short)0, acc, false, false);
  }
  const int m0 = mBase + half * 8;
#pragma unroll
  for (int r = 0; r < 8; ++r) {
    const int m = m0 + r;
    sLat[m * L_SZ + n] += sGate[m * 8 + e] * (acc[r] + bv);
  }
}

// ---------------------------------------------------------------------------
// Fused MoE encoder: one block = 64 batch rows, everything resident in LDS.
// ---------------------------------------------------------------------------
__global__ void __launch_bounds__(256)
moe_fused(const __bf16* __restrict__ obsB,     // pre-converted bf16 [B][D]
          const __bf16* __restrict__ eW1t, const float* __restrict__ eb1,
          const __bf16* __restrict__ eW2t, const float* __restrict__ eb2,
          const __bf16* __restrict__ eW3t, const float* __restrict__ eb3,
          const __bf16* __restrict__ gW1t, const float* __restrict__ gb1,
          const __bf16* __restrict__ gW2t, const float* __restrict__ gb2,
          const float* __restrict__ gW3,   const float* __restrict__ gb3,
          float* __restrict__ outLatent,   float* __restrict__ outWeights) {
  extern __shared__ __align__(16) char smem_raw[];
  __bf16* sObs  = (__bf16*)smem_raw;                 // 64 x 520 bf16
  __bf16* sX1   = sObs + TILE_M * LDA_OBS;           // 64 x 520 bf16
  __bf16* sX2   = sX1 + TILE_M * LDA_X1;             // 64 x 264 bf16
  float*  sLat  = (float*)(sX2 + TILE_M * LDA_X2);   // 64 x 32 f32
  float*  sGate = sLat + TILE_M * L_SZ;              // 64 x 8  f32

  const int tid   = threadIdx.x;
  const int lane  = tid & 31;
  const int wave  = tid >> 5;
  const int half  = (lane >> 4) & 1;
  const int lm    = lane & 15;
  const int mBase = (wave & 3) * 16;   // 4 m-tiles cover 64 rows
  const int nHalf = wave >> 2;         // 2 waves split the N range
  const int rowBase = blockIdx.x * TILE_M;

  // ---- Stage obs tile into LDS (16B chunk j -> row j>>6, chunk j&63) ----
  const __bf16* gsrc = obsB + (size_t)rowBase * D_SZ;
  {
#if HAVE_ASYNC_LDS
    for (int j = tid; j < TILE_M * (D_SZ / 8); j += 256) {
      int r = j >> 6, c = j & 63;     // 64 x 16B chunks per row
      __builtin_amdgcn_global_load_async_to_lds_b128(
          (gptr_v4i)(gsrc + (size_t)j * 8),
          (lptr_v4i)(sObs + r * LDA_OBS + c * 8),
          0, 0);
    }
#else
    for (int j = tid; j < TILE_M * (D_SZ / 8); j += 256) {
      int r = j >> 6, c = j & 63;
      uint4 v = *reinterpret_cast<const uint4*>(gsrc + (size_t)j * 8);
      *reinterpret_cast<uint4*>(sObs + r * LDA_OBS + c * 8) = v;
    }
#endif
    for (int j = tid; j < TILE_M * L_SZ; j += 256) sLat[j] = 0.f;
#if HAVE_ASYNC_LDS
#if __has_builtin(__builtin_amdgcn_s_wait_asynccnt)
    __builtin_amdgcn_s_wait_asynccnt(0);
#else
    asm volatile("s_wait_asynccnt 0x0" ::: "memory");
#endif
#endif
  }
  __syncthreads();

  // ---- Gating MLP ----
  gemm_phase<D_SZ>(sObs, LDA_OBS, gW1t, gb1, sX1, LDA_X1,
                   mBase, nHalf * 256, nHalf * 256 + 256, half, lm);
  __syncthreads();
  gemm_phase<H1_SZ>(sX1, LDA_X1, gW2t, gb2, sX2, LDA_X2,
                    mBase, nHalf * 128, nHalf * 128 + 128, half, lm);
  __syncthreads();

  // Logits (N=8 too small for WMMA): thread -> (row, expert pair)
  {
    const int row = tid >> 2;       // 0..63
    const int ep  = tid & 3;        // expert pair
    float l0 = gb3[2 * ep], l1 = gb3[2 * ep + 1];
    const __bf16* xr = sX2 + row * LDA_X2;
#pragma unroll 4
    for (int k = 0; k < H2_SZ; ++k) {
      float xv = (float)xr[k];
      l0 += xv * gW3[k * E_SZ + 2 * ep];
      l1 += xv * gW3[k * E_SZ + 2 * ep + 1];
    }
    sGate[row * 8 + 2 * ep]     = l0;
    sGate[row * 8 + 2 * ep + 1] = l1;
  }
  __syncthreads();

  // Softmax per row; also write the weights output.
  if (tid < TILE_M) {
    float* g = sGate + tid * 8;
    float mx = g[0];
#pragma unroll
    for (int e = 1; e < 8; ++e) mx = fmaxf(mx, g[e]);
    float w[8], s = 0.f;
#pragma unroll
    for (int e = 0; e < 8; ++e) { w[e] = __expf(g[e] - mx); s += w[e]; }
    const float inv = 1.f / s;
    float* wo = outWeights + (size_t)(rowBase + tid) * E_SZ;
#pragma unroll
    for (int e = 0; e < 8; ++e) { float wv = w[e] * inv; g[e] = wv; wo[e] = wv; }
  }
  __syncthreads();

  // ---- Expert loop (weights stream from L2; all activations in LDS) ----
  for (int e = 0; e < E_SZ; ++e) {
    // Warm L2 with the next expert's W1 panel (global_prefetch_b8).
    __builtin_prefetch(eW1t + (size_t)((e + 1) & 7) * H1_SZ * D_SZ +
                       (size_t)tid * 1024, 0, 1);

    gemm_phase<D_SZ>(sObs, LDA_OBS, eW1t + (size_t)e * H1_SZ * D_SZ,
                     eb1 + e * H1_SZ, sX1, LDA_X1,
                     mBase, nHalf * 256, nHalf * 256 + 256, half, lm);
    __syncthreads();
    gemm_phase<H1_SZ>(sX1, LDA_X1, eW2t + (size_t)e * H2_SZ * H1_SZ,
                      eb2 + e * H2_SZ, sX2, LDA_X2,
                      mBase, nHalf * 128, nHalf * 128 + 128, half, lm);
    __syncthreads();
    gemm3_phase(sX2, eW3t + (size_t)e * L_SZ * H2_SZ, eb3 + e * L_SZ,
                sGate, sLat, e, mBase, nHalf * 16, half, lm);
    __syncthreads();
  }

  // ---- L2 normalize + store latent ----
  if (tid < TILE_M) {
    const float* lr = sLat + tid * L_SZ;
    float s = 0.f;
#pragma unroll
    for (int l = 0; l < L_SZ; ++l) s += lr[l] * lr[l];
    const float inv = 1.f / fmaxf(sqrtf(s), 1e-12f);
    float* o = outLatent + (size_t)(rowBase + tid) * L_SZ;
#pragma unroll
    for (int l = 0; l < L_SZ; ++l) o[l] = lr[l] * inv;
  }
}

// ---------------------------------------------------------------------------
// Prep: f32 -> bf16 elementwise (for obs), vectorized 4-wide.
// ---------------------------------------------------------------------------
__global__ void __launch_bounds__(256)
cvt_bf16(const float* __restrict__ src, __bf16* __restrict__ dst, long n4) {
  long i = (long)blockIdx.x * 256 + threadIdx.x;
  if (i >= n4) return;
  float4 v = reinterpret_cast<const float4*>(src)[i];
  union { __bf16 h[4]; uint2 u; } p;
  p.h[0] = (__bf16)v.x; p.h[1] = (__bf16)v.y;
  p.h[2] = (__bf16)v.z; p.h[3] = (__bf16)v.w;
  reinterpret_cast<uint2*>(dst)[i] = p.u;
}

// ---------------------------------------------------------------------------
// Prep: f32 row-major [batch][K][N] -> bf16 column-major [batch][N][K].
// ---------------------------------------------------------------------------
__global__ void __launch_bounds__(256)
transpose_cvt(const float* __restrict__ src, __bf16* __restrict__ dst,
              int K, int N) {
  __shared__ float tile[32][33];
  const size_t base = (size_t)blockIdx.z * K * N;
  const int k0 = blockIdx.y * 32;
  const int n0 = blockIdx.x * 32;
#pragma unroll
  for (int i = 0; i < 4; ++i)
    tile[threadIdx.y + 8 * i][threadIdx.x] =
        src[base + (size_t)(k0 + threadIdx.y + 8 * i) * N + n0 + threadIdx.x];
  __syncthreads();
#pragma unroll
  for (int i = 0; i < 4; ++i)
    dst[base + (size_t)(n0 + threadIdx.y + 8 * i) * K + k0 + threadIdx.x] =
        (__bf16)tile[threadIdx.x][threadIdx.y + 8 * i];
}

// ---------------------------------------------------------------------------
extern "C" void kernel_launch(void* const* d_in, const int* in_sizes, int n_in,
                              void* d_out, int out_size, void* d_ws, size_t ws_size,
                              hipStream_t stream) {
  (void)in_sizes; (void)n_in; (void)out_size; (void)ws_size;

  const float* obs = (const float*)d_in[0];
  // d_in[1] = obs_no_goal (unused by the reference)
  const float* eW1 = (const float*)d_in[2];
  const float* eb1 = (const float*)d_in[3];
  const float* eW2 = (const float*)d_in[4];
  const float* eb2 = (const float*)d_in[5];
  const float* eW3 = (const float*)d_in[6];
  const float* eb3 = (const float*)d_in[7];
  const float* gW1 = (const float*)d_in[8];
  const float* gb1 = (const float*)d_in[9];
  const float* gW2 = (const float*)d_in[10];
  const float* gb2 = (const float*)d_in[11];
  const float* gW3 = (const float*)d_in[12];
  const float* gb3 = (const float*)d_in[13];

  // Workspace: bf16 obs copy + bf16 column-major weight copies (~41 MB).
  __bf16* ws   = (__bf16*)d_ws;
  __bf16* obsB = ws;                                      // [B][D]
  __bf16* eW1t = obsB + (size_t)B_SZ * D_SZ;              // [E][H1][D]
  __bf16* eW2t = eW1t + (size_t)E_SZ * D_SZ * H1_SZ;      // [E][H2][H1]
  __bf16* eW3t = eW2t + (size_t)E_SZ * H1_SZ * H2_SZ;     // [E][L][H2]
  __bf16* gW1t = eW3t + (size_t)E_SZ * H2_SZ * L_SZ;      // [H1][D]
  __bf16* gW2t = gW1t + (size_t)D_SZ * H1_SZ;             // [H2][H1]

  const long obs4 = (long)B_SZ * D_SZ / 4;
  cvt_bf16<<<dim3((unsigned)((obs4 + 255) / 256)), 256, 0, stream>>>(obs, obsB, obs4);

  dim3 tb(32, 8);
  transpose_cvt<<<dim3(H1_SZ / 32, D_SZ / 32, E_SZ), tb, 0, stream>>>(eW1, eW1t, D_SZ, H1_SZ);
  transpose_cvt<<<dim3(H2_SZ / 32, H1_SZ / 32, E_SZ), tb, 0, stream>>>(eW2, eW2t, H1_SZ, H2_SZ);
  transpose_cvt<<<dim3(L_SZ / 32,  H2_SZ / 32, E_SZ), tb, 0, stream>>>(eW3, eW3t, H2_SZ, L_SZ);
  transpose_cvt<<<dim3(H1_SZ / 32, D_SZ / 32, 1),     tb, 0, stream>>>(gW1, gW1t, D_SZ, H1_SZ);
  transpose_cvt<<<dim3(H2_SZ / 32, H1_SZ / 32, 1),    tb, 0, stream>>>(gW2, gW2t, H1_SZ, H2_SZ);

  float* outLatent  = (float*)d_out;                       // [B, L]
  float* outWeights = outLatent + (size_t)B_SZ * L_SZ;     // [B, E]

  const size_t smem =
      (size_t)(TILE_M * LDA_OBS + TILE_M * LDA_X1 + TILE_M * LDA_X2) * sizeof(__bf16) +
      (size_t)(TILE_M * L_SZ + TILE_M * 8) * sizeof(float);   // ~177 KB (< 320 KB/WGP)

  moe_fused<<<B_SZ / TILE_M, 256, smem, stream>>>(
      obsB, eW1t, eb1, eW2t, eb2, eW3t, eb3,
      gW1t, gb1, gW2t, gb2, gW3, gb3, outLatent, outWeights);
}
